// Q_JetTagger_24756191494277
// MI455X (gfx1250) — 8-bit matrix operand order repaired and checked by reading, compiled, not run on hardware
//
#include <hip/hip_runtime.h>
#include <hip/hip_bf16.h>
#include <math.h>


typedef __attribute__((ext_vector_type(8))) int v8i;

#define ACT_STRIDE 72

struct Ws {
  unsigned int maxabs[4];
  float wscale1[64], wscale2[32], wscale3[32], wscale4[16];
  float bias1[64],  bias2[32],  bias3[32],  bias4[16];
  signed char w1[64 * 64];
  signed char w2[32 * 64];
  signed char w3[32 * 64];
  signed char w4[16 * 64];
};

__device__ __forceinline__ float scale_from_bits(unsigned bits) {
  return fmaxf(__uint_as_float(bits) * (1.0f / 32767.0f), 1e-8f);
}

enum OutMode { OUT_REQUANT = 0, OUT_MAXABS = 1, OUT_RAW = 2 };

template <int NOUT, bool RELU, int MODE>
__device__ __forceinline__ void layer_tile(
    unsigned short* act,
    const signed char* wq,
    const float* bsc,
    const float* bfl,
    float inv_next_scale,
    float* maxabs_acc,
    float* rawout,
    int lane)
{
  const int m    = lane & 15;
  const int half = lane >> 4;

  v8i A_lo = 0, A_hi = 0;
  {
    const unsigned short* row = act + m * ACT_STRIDE;
#pragma unroll
    for (int g = 0; g < 4; ++g) {
      const uint4 q = *(const uint4*)(row + (g >> 1) * 32 + half * 16 + (g & 1) * 8);
      A_lo[2 * g]     = (int)__builtin_amdgcn_perm(q.y, q.x, 0x06040200u);
      A_hi[2 * g]     = (int)__builtin_amdgcn_perm(q.y, q.x, 0x07050301u);
      A_lo[2 * g + 1] = (int)__builtin_amdgcn_perm(q.w, q.z, 0x06040200u);
      A_hi[2 * g + 1] = (int)__builtin_amdgcn_perm(q.w, q.z, 0x07050301u);
    }
  }

  constexpr int NT = (NOUT + 15) / 16;
#pragma unroll
  for (int t = 0; t < NT; ++t) {
    const int n = t * 16 + m;

    const signed char* wrow = wq + n * 64;
    const uint4 blo = *(const uint4*)(wrow + half * 16);
    const uint4 bhi = *(const uint4*)(wrow + 32 + half * 16);
    v8i B;
    B[0] = (int)blo.x; B[1] = (int)blo.y; B[2] = (int)blo.z; B[3] = (int)blo.w;
    B[4] = (int)bhi.x; B[5] = (int)bhi.y; B[6] = (int)bhi.z; B[7] = (int)bhi.w;

    v8i acc_hi = 0, acc_lo = 0;
    acc_hi = __builtin_amdgcn_wmma_i32_16x16x64_iu8(true,  A_hi, true, B, acc_hi, false, false);
    acc_lo = __builtin_amdgcn_wmma_i32_16x16x64_iu8(false, A_lo, true, B, acc_lo, false, false);

    const float sc = bsc[n];
    const float bf = bfl[n];

#pragma unroll
    for (int r = 0; r < 8; ++r) {
      const int mo  = r + (half << 3);
      const int res = acc_hi[r] * 256 + acc_lo[r];
      float h = fmaf((float)res, sc, bf);
      if (RELU) h = fmaxf(h, 0.0f);

      if (MODE == OUT_MAXABS) {
        *maxabs_acc = fmaxf(*maxabs_acc, h);
      } else if (MODE == OUT_REQUANT) {
        const float q = fminf(fmaxf(rintf(h * inv_next_scale), -32768.0f), 32767.0f);
        act[mo * ACT_STRIDE + n] = (unsigned short)(short)(int)q;
      } else {
        if (n < 8) rawout[mo * 8 + n] = h;
      }
    }
  }

  if (MODE == OUT_REQUANT) {
#pragma unroll
    for (int t = NT; t < 4; ++t) {
      const int n = t * 16 + m;
#pragma unroll
      for (int r = 0; r < 8; ++r)
        act[(r + (half << 3)) * ACT_STRIDE + n] = 0;
    }
  }
}

#define CH_L1 0
#define CH_L2 64
#define CH_L3 96
#define CH_L4 128
#define CH_TOT 144

template <int UPTO>
__global__ __launch_bounds__(128) void mlp_pass(
    const float* __restrict__ x, Ws* __restrict__ ws, float* __restrict__ out)
{
  __shared__ alignas(16) Ws wss;
  __shared__ alignas(16) unsigned short act_s[4][16][ACT_STRIDE];
  __shared__ float g_bsc[CH_TOT], g_bfl[CH_TOT];

  {
    const uint4* src = (const uint4*)ws;
    uint4* dst = (uint4*)&wss;
    for (int i = threadIdx.x; i < (int)(sizeof(Ws) / 16); i += (int)blockDim.x) dst[i] = src[i];
  }
  __syncthreads();

  const float s1 = scale_from_bits(wss.maxabs[0]);
  const float s2 = (UPTO >= 2) ? scale_from_bits(wss.maxabs[1]) : 1.0f;
  const float s3 = (UPTO >= 3) ? scale_from_bits(wss.maxabs[2]) : 1.0f;
  const float s4 = (UPTO >= 4) ? scale_from_bits(wss.maxabs[3]) : 1.0f;

  for (int i = threadIdx.x; i < CH_TOT; i += (int)blockDim.x) {
    float sl, wc, bc;
    if (i < CH_L2)      { sl = s1; wc = wss.wscale1[i];           bc = wss.bias1[i]; }
    else if (i < CH_L3) { sl = s2; wc = wss.wscale2[i - CH_L2];   bc = wss.bias2[i - CH_L2]; }
    else if (i < CH_L4) { sl = s3; wc = wss.wscale3[i - CH_L3];   bc = wss.bias3[i - CH_L3]; }
    else                { sl = s4; wc = wss.wscale4[i - CH_L4];   bc = wss.bias4[i - CH_L4]; }
    const float sc = wc * sl;
    g_bsc[i] = sc;
    g_bfl[i] = rintf(bc / sc) * sc;
  }

  const int lane = threadIdx.x & 31;
  const int wave = threadIdx.x >> 5;
  unsigned short* act = &act_s[wave][0][0];
  const long rowbase = ((long)blockIdx.x * 4 + wave) * 16;

#pragma unroll
  for (int i = 0; i < (16 * ACT_STRIDE / 2) / 32; ++i) ((unsigned*)act)[lane + i * 32] = 0u;
  {
    const float inv1 = 1.0f / s1;
    const int r  = lane >> 1;
    const int c0 = (lane & 1) * 8;
    const float* xr = x + (rowbase + r) * 16 + c0;
#pragma unroll
    for (int j = 0; j < 8; j += 2) {
      const float qa = fminf(fmaxf(rintf(xr[j]     * inv1), -32768.0f), 32767.0f);
      const float qb = fminf(fmaxf(rintf(xr[j + 1] * inv1), -32768.0f), 32767.0f);
      const unsigned pa = (unsigned)(unsigned short)(short)(int)qa;
      const unsigned pb = (unsigned)(unsigned short)(short)(int)qb;
      *(unsigned*)(act + r * ACT_STRIDE + c0 + j) = pa | (pb << 16);
    }
  }
  __syncthreads();

  float mx = 0.0f;
  if constexpr (UPTO == 1) {
    layer_tile<64, true, OUT_MAXABS>(act, wss.w1, g_bsc + CH_L1, g_bfl + CH_L1, 0.f, &mx, nullptr, lane);
  } else {
    layer_tile<64, true, OUT_REQUANT>(act, wss.w1, g_bsc + CH_L1, g_bfl + CH_L1, 1.0f / s2, nullptr, nullptr, lane);
    __syncthreads();
    if constexpr (UPTO == 2) {
      layer_tile<32, true, OUT_MAXABS>(act, wss.w2, g_bsc + CH_L2, g_bfl + CH_L2, 0.f, &mx, nullptr, lane);
    } else {
      layer_tile<32, true, OUT_REQUANT>(act, wss.w2, g_bsc + CH_L2, g_bfl + CH_L2, 1.0f / s3, nullptr, nullptr, lane);
      __syncthreads();
      if constexpr (UPTO == 3) {
        layer_tile<32, true, OUT_MAXABS>(act, wss.w3, g_bsc + CH_L3, g_bfl + CH_L3, 0.f, &mx, nullptr, lane);
      } else {
        layer_tile<32, true, OUT_REQUANT>(act, wss.w3, g_bsc + CH_L3, g_bfl + CH_L3, 1.0f / s4, nullptr, nullptr, lane);
        __syncthreads();
        float* raw = (float*)act;
        layer_tile<5, false, OUT_RAW>(act, wss.w4, g_bsc + CH_L4, g_bfl + CH_L4, 0.f, nullptr, raw, lane);
        __syncthreads();
        if (lane < 16) {
          const float v0 = raw[lane * 8 + 0], v1 = raw[lane * 8 + 1], v2 = raw[lane * 8 + 2];
          const float v3 = raw[lane * 8 + 3], v4 = raw[lane * 8 + 4];
          const float mv = fmaxf(fmaxf(fmaxf(v0, v1), fmaxf(v2, v3)), v4);
          const float e0 = __expf(v0 - mv), e1 = __expf(v1 - mv), e2 = __expf(v2 - mv);
          const float e3 = __expf(v3 - mv), e4 = __expf(v4 - mv);
          const float inv = 1.0f / (e0 + e1 + e2 + e3 + e4);
          float* o = out + (rowbase + lane) * 5;
          o[0] = e0 * inv; o[1] = e1 * inv; o[2] = e2 * inv; o[3] = e3 * inv; o[4] = e4 * inv;
        }
      }
    }
  }

  if constexpr (UPTO <= 3) {
#pragma unroll
    for (int off = 16; off > 0; off >>= 1) mx = fmaxf(mx, __shfl_xor(mx, off, 32));
    if (lane == 0) atomicMax(&ws->maxabs[UPTO], __float_as_uint(mx));
  }
}

__global__ void maxabs_x_kernel(const float* __restrict__ x, long n, unsigned* __restrict__ stat) {
  long i = (long)blockIdx.x * blockDim.x + threadIdx.x;
  const long stride = (long)gridDim.x * blockDim.x;
  float m = 0.0f;
  for (; i < n; i += stride) m = fmaxf(m, fabsf(x[i]));
#pragma unroll
  for (int off = 16; off > 0; off >>= 1) m = fmaxf(m, __shfl_xor(m, off, 32));
  if ((threadIdx.x & 31) == 0) atomicMax(stat, __float_as_uint(m));
}

__device__ __forceinline__ void quant_row_pad(const float* w, int K, signed char* q,
                                              float* sc, float bval, float* bdst) {
  float ma = 0.0f;
  for (int k = 0; k < K; ++k) ma = fmaxf(ma, fabsf(w[k]));
  const float s = fmaxf(ma * (1.0f / 127.0f), 1e-8f);
  *sc = s; *bdst = bval;
  const float inv = 1.0f / s;
  for (int k = 0; k < 64; ++k) {
    float v = 0.0f;
    if (k < K) v = fminf(fmaxf(rintf(w[k] * inv), -128.0f), 127.0f);
    q[k] = (signed char)(int)v;
  }
}

__global__ void prep_kernel(const float* __restrict__ W1, const float* __restrict__ W2,
                            const float* __restrict__ W3, const float* __restrict__ W4,
                            const float* __restrict__ B1, const float* __restrict__ B2,
                            const float* __restrict__ B3, const float* __restrict__ B4,
                            Ws* __restrict__ ws) {
  const int t = threadIdx.x;
  if (t < 4) ws->maxabs[t] = 0u;
  if (t < 64) {
    quant_row_pad(W1 + t * 16, 16, ws->w1 + t * 64, &ws->wscale1[t], B1[t], &ws->bias1[t]);
  } else if (t < 96) {
    const int r = t - 64;
    quant_row_pad(W2 + r * 64, 64, ws->w2 + r * 64, &ws->wscale2[r], B2[r], &ws->bias2[r]);
  } else if (t < 128) {
    const int r = t - 96;
    quant_row_pad(W3 + r * 32, 32, ws->w3 + r * 64, &ws->wscale3[r], B3[r], &ws->bias3[r]);
  } else if (t < 144) {
    const int r = t - 128;
    if (r < 5) {
      quant_row_pad(W4 + r * 32, 32, ws->w4 + r * 64, &ws->wscale4[r], B4[r], &ws->bias4[r]);
    } else {
      ws->wscale4[r] = 1.0f; ws->bias4[r] = 0.0f;
      for (int k = 0; k < 64; ++k) ws->w4[r * 64 + k] = 0;
    }
  }
}

extern "C" void kernel_launch(void* const* d_in, const int* in_sizes, int n_in,
                              void* d_out, int out_size, void* d_ws, size_t ws_size,
                              hipStream_t stream) {
  (void)n_in; (void)out_size; (void)ws_size;
  const float* x  = (const float*)d_in[0];
  const float* W1 = (const float*)d_in[1]; const float* b1 = (const float*)d_in[2];
  const float* W2 = (const float*)d_in[3]; const float* b2 = (const float*)d_in[4];
  const float* W3 = (const float*)d_in[5]; const float* b3 = (const float*)d_in[6];
  const float* W4 = (const float*)d_in[7]; const float* b4 = (const float*)d_in[8];
  Ws* ws = (Ws*)d_ws;
  float* out = (float*)d_out;

  const long nx = (long)in_sizes[0];
  const long B  = nx / 16;
  const int blocks = (int)(B / 64);

  prep_kernel<<<1, 256, 0, stream>>>(W1, W2, W3, W4, b1, b2, b3, b4, ws);
  maxabs_x_kernel<<<2048, 256, 0, stream>>>(x, nx, &ws->maxabs[0]);
  mlp_pass<1><<<blocks, 128, 0, stream>>>(x, ws, out);
  mlp_pass<2><<<blocks, 128, 0, stream>>>(x, ws, out);
  mlp_pass<3><<<blocks, 128, 0, stream>>>(x, ws, out);
  mlp_pass<4><<<blocks, 128, 0, stream>>>(x, ws, out);
}
